// Network_37924561224237
// MI455X (gfx1250) — matrix operand order repaired and checked by reading, not compiled, not run on hardware
//
#include <hip/hip_runtime.h>

// ---------------------------------------------------------------------------
// MI455X (gfx1250) implementation.
//
// The reference network is entirely linear (no activations; dropout is eval-
// mode identity) and both GCN layers share the same normalized adjacency A
// (same edge list, same degrees). The MLP tail collapses exactly:
//     out = x1 . weff + c,  weff = (Wl1@Wl2@Wl3 + I + Wl1) @ Wl4  (256-vec)
// and folding weff's four 64-blocks back through W2 and W1 collapses the
// whole network, per graph, to
//     S[p][n] = x[p] . U[n]            (4x4, U[n] = W1@(W2@w_n) is [14])
//     out     = trace(A*A*S) + sum_n beta[n]*rowsum(A)[n] + C0
// FLOPs drop from ~125 GFLOP (compute-bound even with WMMA, ~25us) to
// ~0.3 GFLOP; the kernel is a pure stream of x (58.7MB) + edges (25.2MB)
// -> ~85MB / 23.3TB/s ~= 3.6us.  Data movement uses the Tensor Data Mover
// (tensor_load_to_lds, TENSORcnt) with a hand-built D# (ISA ch.8); the
// once-per-launch precompute uses v_wmma_f32_16x16x32_f16 (f32 acc).
// ---------------------------------------------------------------------------

typedef __attribute__((ext_vector_type(16))) _Float16 v16h;
typedef __attribute__((ext_vector_type(8)))  float    v8f;

#define NNODES 4
#define NEDGES 12
#define GPB    128   // graphs per block (main kernel)
#define TPB    128   // threads per block (4 wave32)

// ---------------------------------------------------------------------------
// Tensor Data Mover: 1-D dword copy global -> LDS via a hand-built D#.
//   tile_dim0   = slab dwords (LDS always fully written)
//   tensor_dim0 = valid dwords (hardware zero-fills OOB reads on tail blocks)
// Issued by one wave per block; completion via s_wait_tensorcnt.
// This toolchain's builtin is the 6-operand form (probe: clang-23/therock):
//   (uint32x4 g0, int32x8 g1, int32x4, int32x4, int32x8, i32 cpol)
// Groups 2/3 and the extended-dim words are zero: 1-D tile, dims 2..4 unused.
// ---------------------------------------------------------------------------
#if __has_builtin(__builtin_amdgcn_tensor_load_to_lds) && \
    __has_builtin(__builtin_amdgcn_s_wait_tensorcnt)
#define HAVE_TDM 1
typedef __attribute__((ext_vector_type(4))) unsigned int tdm_u32x4;
typedef __attribute__((ext_vector_type(8))) int          tdm_i32x8;
typedef __attribute__((ext_vector_type(4))) int          tdm_i32x4;

__device__ __forceinline__ void tdm_copy_1d(const void* gsrc,
                                            unsigned lds_addr,
                                            unsigned tile_dwords,
                                            unsigned valid_dwords)
{
  const unsigned long long ga = (unsigned long long)gsrc;
  tdm_u32x4 g0;
  g0[0] = 1u;                                    // count=1, user mode, no gather
  g0[1] = lds_addr;                              // LDS byte address [63:32]
  g0[2] = (unsigned)(ga & 0xFFFFFFFFu);          // global_addr[31:0]
  g0[3] = (unsigned)((ga >> 32) & 0x01FFFFFFu)   // global_addr[56:32]
          | (2u << 30);                          // type=2 ("image")
  tdm_i32x8 g1;
  g1[0] = (int)(2u << 16);                       // wg_mask=0, data_size=4B
  g1[1] = (int)((valid_dwords & 0xFFFFu) << 16); // tensor_dim0[15:0]
  g1[2] = (int)((valid_dwords >> 16) & 0xFFFFu)  // tensor_dim0[31:16]
          | (1 << 16);                           // tensor_dim1 = 1
  g1[3] = (int)((tile_dwords & 0xFFFFu) << 16);  // tile_dim0
  g1[4] = 0;                                     // tile_dim1=0, tile_dim2=0 (1D)
  g1[5] = (int)tile_dwords;                      // tensor_dim0_stride
  g1[6] = 0;
  g1[7] = 0;
  const tdm_i32x4 gz4 = {0, 0, 0, 0};            // dims 3/4 unused
  const tdm_i32x8 gz8 = {0, 0, 0, 0, 0, 0, 0, 0};
  __builtin_amdgcn_tensor_load_to_lds(g0, g1, gz4, gz4, gz8, 0);
}
#else
#define HAVE_TDM 0
#endif

// ---------------------------------------------------------------------------
// Precompute kernel: one block of 256 threads (8 waves).
//   ws[0..55]  = U[n][k]  (n-major, n<4, k<14)
//   ws[56..59] = beta[n]
//   ws[60]     = C0 (all bias contributions)
// ---------------------------------------------------------------------------
__global__ __launch_bounds__(256) void precompute_kernel(
    const float* __restrict__ W1,  const float* __restrict__ b1,
    const float* __restrict__ W2,  const float* __restrict__ b2,
    const float* __restrict__ Wl1, const float* __restrict__ bl1,
    const float* __restrict__ Wl2, const float* __restrict__ bl2,
    const float* __restrict__ Wl3, const float* __restrict__ bl3,
    const float* __restrict__ Wl4, const float* __restrict__ bl4,
    float* __restrict__ ws)
{
  __shared__ float s_w4[256], s_t3[256], s_t2[256], s_weff[256];
  __shared__ float s_V[128 * 4];
  __shared__ float s_c;
  const int tid = threadIdx.x;

  // w4 = Wl4  (shape [256,1])
  s_w4[tid] = Wl4[tid];
  __syncthreads();

  // t3 = Wl3 @ w4   (exact fp32 matvec; trivial cost)
  {
    const float* row = Wl3 + tid * 256;
    float a = 0.f;
    #pragma unroll 8
    for (int j = 0; j < 256; ++j) a += row[j] * s_w4[j];
    s_t3[tid] = a;
  }
  __syncthreads();

  // t2 = Wl2 @ t3
  {
    const float* row = Wl2 + tid * 256;
    float a = 0.f;
    #pragma unroll 8
    for (int j = 0; j < 256; ++j) a += row[j] * s_t3[j];
    s_t2[tid] = a;
  }
  __syncthreads();

  // weff = Wl1@t2 + w4 + Wl1@w4 ;  scalar bias constant c
  {
    const float* row = Wl1 + tid * 256;
    float t1 = 0.f, w1w4 = 0.f;
    #pragma unroll 8
    for (int j = 0; j < 256; ++j) {
      t1   += row[j] * s_t2[j];
      w1w4 += row[j] * s_w4[j];
    }
    s_weff[tid] = t1 + s_w4[tid] + w1w4;
  }
  if (tid == 0) {
    float c = bl4[0];
    for (int i = 0; i < 256; ++i)
      c += bl1[i] * s_t2[i] + bl2[i] * s_t3[i] + (bl3[i] + bl1[i]) * s_w4[i];
    s_c = c;
  }
  __syncthreads();

  // ---- V[i][n] = sum_j W2[i*64+j] * weff[64n+j]  via WMMA ---------------
  // GEMM: C[128,4] = A[128,64] @ B[64,4]; 8 waves, one 16-row tile each,
  // K = 64 -> two 16x16x32 f16 WMMA steps, f32 accumulate.
  // Operand packing uses clamped indices * 0/1 mask (no divergent loads).
  const int wave = tid >> 5;
  const int lane = tid & 31;
  const int hi16 = lane >> 4;   // lane half (K-group select for A/B layouts)
  const int nn   = lane & 15;   // M index for A; N index for B/C/D
  const int   bn    = nn & 3;
  const float bmask = (nn < 4) ? 1.f : 0.f;

  {
    v8f acc = {0.f, 0.f, 0.f, 0.f, 0.f, 0.f, 0.f, 0.f};
    #pragma unroll
    for (int k0 = 0; k0 < 64; k0 += 32) {
      v16h a, b;
      #pragma unroll
      for (int p = 0; p < 8; ++p) {
        #pragma unroll
        for (int jj = 0; jj < 2; ++jj) {
          // ISA 7.12.2: 16-bit A 16x32 per-lane K pattern
          const int kA = (p < 4 ? 2 * p : 16 + 2 * (p - 4)) + 8 * hi16 + jj;
          a[2 * p + jj] = (_Float16)W2[(16 * wave + nn) * 64 + (k0 + kA)];
          // B 32x16: same per-lane K pattern as A
          const int kB = (p < 4 ? 2 * p : 16 + 2 * (p - 4)) + 8 * hi16 + jj;
          b[2 * p + jj] = (_Float16)(s_weff[64 * bn + (k0 + kB)] * bmask);
        }
      }
      acc = __builtin_amdgcn_wmma_f32_16x16x32_f16(
          false, a, false, b, (short)0, acc, false, false);
    }
    // D layout: VGPR r -> row r (lanes 0-15) / r+8 (lanes 16-31), col = nn
    if (nn < 4) {
      #pragma unroll
      for (int r = 0; r < 8; ++r)
        s_V[(16 * wave + r + 8 * hi16) * 4 + nn] = acc[r];
    }
  }
  __syncthreads();

  // ---- U[n][k] = sum_i W1[k*128+i] * V[i][n]  via WMMA (wave 0) ---------
  // GEMM: C[14(pad16),4(pad16)] = A[14,128] @ B[128,4]; K=128 -> 4 steps.
  if (wave == 0) {
    const int   am    = (nn < 14) ? nn : 0;
    const float amask = (nn < 14) ? 1.f : 0.f;
    v8f u = {0.f, 0.f, 0.f, 0.f, 0.f, 0.f, 0.f, 0.f};
    #pragma unroll
    for (int k0 = 0; k0 < 128; k0 += 32) {
      v16h a, b;
      #pragma unroll
      for (int p = 0; p < 8; ++p) {
        #pragma unroll
        for (int jj = 0; jj < 2; ++jj) {
          const int kA = (p < 4 ? 2 * p : 16 + 2 * (p - 4)) + 8 * hi16 + jj;
          a[2 * p + jj] = (_Float16)(W1[am * 128 + (k0 + kA)] * amask);
          const int kB = (p < 4 ? 2 * p : 16 + 2 * (p - 4)) + 8 * hi16 + jj;
          b[2 * p + jj] = (_Float16)(s_V[(k0 + kB) * 4 + bn] * bmask);
        }
      }
      u = __builtin_amdgcn_wmma_f32_16x16x32_f16(
          false, a, false, b, (short)0, u, false, false);
    }
    if (nn < 4) {
      #pragma unroll
      for (int r = 0; r < 8; ++r) {
        const int m = r + 8 * hi16;
        if (m < 14) ws[nn * 14 + m] = u[r];
      }
    }
  }

  // beta[n] = b1 . V[:,n] ; C0 = c + sum_n b2 . w_n
  if (tid < 4) {
    float be = 0.f;
    for (int i = 0; i < 128; ++i) be += b1[i] * s_V[i * 4 + tid];
    ws[56 + tid] = be;
  } else if (tid == 4) {
    float g = s_c;
    for (int n = 0; n < 4; ++n)
      for (int j = 0; j < 64; ++j) g += b2[j] * s_weff[64 * n + j];
    ws[60] = g;
  }
}

// ---------------------------------------------------------------------------
// Main streaming kernel: one graph per lane, 128 graphs per block.
// Slab staging via TDM (tensor_load_to_lds + s_wait_tensorcnt) when
// available, else coalesced float4/int4 loads.  Per-graph math fully in
// registers (static indexing only); coefficient reads are uniform -> SGPR.
// ---------------------------------------------------------------------------
__device__ __forceinline__ float sel4(float a0, float a1, float a2, float a3,
                                      int i) {
  float r = a0;
  r = (i == 1) ? a1 : r;
  r = (i == 2) ? a2 : r;
  r = (i == 3) ? a3 : r;
  return r;
}

__global__ __launch_bounds__(TPB) void gcn_mlp_fused_kernel(
    const float* __restrict__ x, const int* __restrict__ edge_index,
    const float* __restrict__ coef, float* __restrict__ out, int nB)
{
  __shared__ float4 xs4[GPB * 14];  // 56 floats per graph (28 KB)
  __shared__ int4   es4[GPB * 6];   // 24 ints  per graph (12 KB)
  const int tid = threadIdx.x;
  const int g0  = blockIdx.x * GPB;

#if HAVE_TDM
  // One wave issues two TDM descriptors; OOB (tail block) reads zero-fill.
  if (tid < 32) {
    const int gcnt = (nB - g0 < GPB) ? (nB - g0) : GPB;
    tdm_copy_1d(x + (size_t)g0 * 56,
                (unsigned)(unsigned long long)(const void*)xs4,
                GPB * 56u, (unsigned)(gcnt * 56));
    tdm_copy_1d(edge_index + (size_t)g0 * 24,
                (unsigned)(unsigned long long)(const void*)es4,
                GPB * 24u, (unsigned)(gcnt * 24));
    __builtin_amdgcn_s_wait_tensorcnt(0);
  }
  __syncthreads();
#else
  const float4* xp = (const float4*)x;
  const int4*   ep = (const int4*)edge_index;
  const int xbase = g0 * 14, ebase = g0 * 6;
  const int xtot  = nB * 14, etot  = nB * 6;
  {
    const char* pf = (const char*)(xp + xbase);
    __builtin_prefetch(pf + tid * 224, 0, 0);
    __builtin_prefetch(pf + tid * 224 + 128, 0, 0);
  }
  #pragma unroll
  for (int i = tid; i < GPB * 6; i += TPB) {
    const int gi = ebase + i;
    es4[i] = (gi < etot) ? ep[gi] : make_int4(0, 0, 0, 0);
  }
  #pragma unroll
  for (int i = tid; i < GPB * 14; i += TPB) {
    const int gi = xbase + i;
    xs4[i] = (gi < xtot) ? xp[gi] : make_float4(0.f, 0.f, 0.f, 0.f);
  }
  __syncthreads();
#endif

  const int g = g0 + tid;
  if (g >= nB) return;

  // ---- edges: first 12 = src, next 12 = dst (edge_index[:,0], [:,1]) ----
  int src[NEDGES], dst[NEDGES];
  {
    const int* eg = (const int*)(es4 + tid * 6);
    #pragma unroll
    for (int e = 0; e < NEDGES; ++e) {
      src[e] = eg[e] & 3;
      dst[e] = eg[NEDGES + e] & 3;
    }
  }

  // ---- degrees (self loop adds 1 to each node) --------------------------
  float dg0 = 1.f, dg1 = 1.f, dg2 = 1.f, dg3 = 1.f;
  #pragma unroll
  for (int e = 0; e < NEDGES; ++e) {
    const int d = dst[e];
    dg0 += (d == 0) ? 1.f : 0.f;
    dg1 += (d == 1) ? 1.f : 0.f;
    dg2 += (d == 2) ? 1.f : 0.f;
    dg3 += (d == 3) ? 1.f : 0.f;
  }
  const float di0 = rsqrtf(dg0), di1 = rsqrtf(dg1);
  const float di2 = rsqrtf(dg2), di3 = rsqrtf(dg3);

  // ---- normalized adjacency A[d][s] (4x4, registers) --------------------
  float A[16];
  #pragma unroll
  for (int i = 0; i < 16; ++i) A[i] = 0.f;
  A[0]  = di0 * di0;  A[5]  = di1 * di1;
  A[10] = di2 * di2;  A[15] = di3 * di3;
  #pragma unroll
  for (int e = 0; e < NEDGES; ++e) {
    const int idx = dst[e] * 4 + src[e];
    const float w = sel4(di0, di1, di2, di3, src[e]) *
                    sel4(di0, di1, di2, di3, dst[e]);
    #pragma unroll
    for (int i = 0; i < 16; ++i) A[i] += (idx == i) ? w : 0.f;
  }

  // ---- S[p][n] = x[p] . U[n]   (coef is uniform -> scalar loads) --------
  float S[16];
  const float* xg = (const float*)(xs4 + tid * 14);
  #pragma unroll
  for (int p = 0; p < 4; ++p) {
    float xr[14];
    #pragma unroll
    for (int k = 0; k < 14; ++k) xr[k] = xg[p * 14 + k];
    #pragma unroll
    for (int n = 0; n < 4; ++n) {
      float a = 0.f;
      #pragma unroll
      for (int k = 0; k < 14; ++k) a += xr[k] * coef[n * 14 + k];
      S[p * 4 + n] = a;
    }
  }

  // ---- T = A @ S ; result = trace(A*T) + beta.rowsum(A) + C0 ------------
  float T[16];
  #pragma unroll
  for (int m = 0; m < 4; ++m)
    #pragma unroll
    for (int n = 0; n < 4; ++n) {
      float a = 0.f;
      #pragma unroll
      for (int p = 0; p < 4; ++p) a += A[m * 4 + p] * S[p * 4 + n];
      T[m * 4 + n] = a;
    }

  float r = 0.f;
  #pragma unroll
  for (int n = 0; n < 4; ++n) {
    float rsn = 0.f;
    #pragma unroll
    for (int m = 0; m < 4; ++m) {
      r   += A[n * 4 + m] * T[m * 4 + n];
      rsn += A[n * 4 + m];
    }
    r += coef[56 + n] * rsn;
  }
  out[g] = r + coef[60];
}

// ---------------------------------------------------------------------------
extern "C" void kernel_launch(void* const* d_in, const int* in_sizes, int n_in,
                              void* d_out, int out_size, void* d_ws,
                              size_t ws_size, hipStream_t stream)
{
  const float* x    = (const float*)d_in[0];
  const int*   edge = (const int*)d_in[1];
  const float* W1   = (const float*)d_in[2];
  const float* b1   = (const float*)d_in[3];
  const float* W2   = (const float*)d_in[4];
  const float* b2   = (const float*)d_in[5];
  const float* Wl1  = (const float*)d_in[6];
  const float* bl1  = (const float*)d_in[7];
  const float* Wl2  = (const float*)d_in[8];
  const float* bl2  = (const float*)d_in[9];
  const float* Wl3  = (const float*)d_in[10];
  const float* bl3  = (const float*)d_in[11];
  const float* Wl4  = (const float*)d_in[12];
  const float* bl4  = (const float*)d_in[13];
  float* out = (float*)d_out;
  float* ws  = (float*)d_ws;

  const int B = in_sizes[0] / (NNODES * 14);   // x is [B, 4, 14]

  precompute_kernel<<<1, 256, 0, stream>>>(W1, b1, W2, b2, Wl1, bl1, Wl2, bl2,
                                           Wl3, bl3, Wl4, bl4, ws);
  const int blocks = (B + GPB - 1) / GPB;
  gcn_mlp_fused_kernel<<<blocks, TPB, 0, stream>>>(x, edge, ws, out, B);
}
